// TorchHawkes_19705309954599
// MI455X (gfx1250) — matrix operand order repaired and checked by reading, not compiled, not run on hardware
//
#include <hip/hip_runtime.h>
#include <hip/hip_bf16.h>
#include <math.h>

typedef __attribute__((ext_vector_type(16))) _Float16 v16h;
typedef __attribute__((ext_vector_type(8)))  float    v8f;

constexpr int Kdim = 256;
constexpr int Ndim = 65536;
constexpr int GEMM_BLOCKS = 512;      // 65536 t-rows / 128 per block

union Frag16 { v16h h; uint4 q[2]; }; // 32 bytes: one WMMA f16 operand per lane

// ---------------------------------------------------------------------------
// Phase 1: Mu0[k] = mean(obs[k,:]) / 10 + 0.01   (one block per row, coalesced)
// ---------------------------------------------------------------------------
__global__ void __launch_bounds__(256) hawkes_mean(const float* __restrict__ obs,
                                                   float* __restrict__ mu0) {
    const int k = blockIdx.x;
    const float* row = obs + (size_t)k * Ndim;
    float s = 0.f;
    for (int t = threadIdx.x; t < Ndim; t += 256) s += row[t];
    __shared__ float red[256];
    red[threadIdx.x] = s;
    __syncthreads();
    for (int st = 128; st > 0; st >>= 1) {
        if (threadIdx.x < st) red[threadIdx.x] += red[threadIdx.x + st];
        __syncthreads();
    }
    if (threadIdx.x == 0) mu0[k] = red[0] * (1.0f / (float)Ndim) * 0.1f + 0.01f;
}

// ---------------------------------------------------------------------------
// Phase 2a: pack Alpha (f32 [j][k]) into fragment-contiguous f16 B operands.
// Fragment f = kt*16 + jt (kt: K-tile of 32, jt: 16-col j-tile).
// Bq[((f*32 + lane)*16) + i] = Alpha[j][K],  K = kt*32 + (lane<16?0:16) + i,
// j = jt*16 + (lane&15)  ->  each lane's fragment = 32 contiguous bytes.
// ---------------------------------------------------------------------------
__global__ void __launch_bounds__(256) hawkes_alpha_pack(const float* __restrict__ Alpha,
                                                         _Float16* __restrict__ Bq) {
    const int idx  = blockIdx.x * 256 + threadIdx.x;  // 0..65535
    const int i    = idx & 15;
    const int lane = (idx >> 4) & 31;
    const int f    = idx >> 9;                        // 0..127
    const int kt   = f >> 4;
    const int jt   = f & 15;
    const int K    = kt * 32 + ((lane < 16) ? 0 : 16) + i;
    const int j    = jt * 16 + (lane & 15);
    Bq[idx] = (_Float16)Alpha[(size_t)j * 256 + K];
}

// ---------------------------------------------------------------------------
// Phase 2b: exponential-kernel scan, time-parallel.
// 64-step warm-up decouples 256-step chunks (e^{-64*Beta} < 2e-28 for Beta>=1,
// below f32 significance). Thread = k -> coalesced f16 S[t][k] stores.
// ---------------------------------------------------------------------------
__global__ void __launch_bounds__(256) hawkes_scan(const float* __restrict__ obs,
                                                   const float* __restrict__ Beta,
                                                   _Float16* __restrict__ S) {
    const int k  = threadIdx.x;
    const int t0 = blockIdx.x * 256;
    const float beta  = Beta[k];
    const float decay = expf(-beta);
    const float* row  = obs + (size_t)k * Ndim;

    float h = 0.f;
    int tw = t0 - 64;
    if (tw < 0) tw = 0;
    for (int t = tw; t < t0; ++t) h = decay * (h + row[t]);

    for (int t = t0; t < t0 + 256; ++t) {
        if ((t & 63) == 0) __builtin_prefetch(row + t + 320, 0, 1);  // global_prefetch_b8
        h = decay * (h + row[t]);
        S[(size_t)t * Kdim + k] = (_Float16)(beta * h);
    }
}

// ---------------------------------------------------------------------------
// Phase 3: WMMA GEMM  lam[t,j] = softplus( sum_k S[t,k] * Alpha[j,k] )
// Block tile: 128 t x 256 j. Each wave: TWO 16-row t-tiles x eight j-tiles
// -> every B fragment load feeds 2 v_wmma; 128 wmma per wave.
// Epilogue transposes D tiles through padded LDS so lams0/lams1 stores and
// obs loads are 64B-contiguous per j-row; fused loglik partial reduction.
// ---------------------------------------------------------------------------
__global__ void __launch_bounds__(256) hawkes_gemm(const _Float16* __restrict__ S,
                                                   const _Float16* __restrict__ Bq,
                                                   const float* __restrict__ mu0,
                                                   const float* __restrict__ obs,
                                                   float* __restrict__ lams0,
                                                   float* __restrict__ lams1,
                                                   float* __restrict__ partials) {
    const int tid  = threadIdx.x;
    const int lane = tid & 31;
    const int wave = tid >> 5;
    const int tt   = wave & 3;        // t-tile index within half-block
    const int jh   = wave >> 2;       // j half (0: j 0..127, 1: j 128..255)
    const int tbase0 = blockIdx.x * 128 + tt * 16;   // first t-tile
    // second t-tile is tbase0 + 64

    const int m      = lane & 15;                 // A row within tile
    const int khalfA = (lane < 16) ? 0 : 16;      // A half-K select
    const int jcol   = lane & 15;                 // B/D column within tile

    v8f zero = {};
    v8f acc[2][8];
#pragma unroll
    for (int h2 = 0; h2 < 2; ++h2)
#pragma unroll
        for (int jt = 0; jt < 8; ++jt) acc[h2][jt] = zero;

    const _Float16* srow0 = S + (size_t)(tbase0 + m) * Kdim;
    const _Float16* srow1 = srow0 + (size_t)64 * Kdim;

#pragma unroll
    for (int kt = 0; kt < 8; ++kt) {
        const int kb = kt * 32;
        Frag16 a0, a1;
        a0.q[0] = *(const uint4*)(srow0 + kb + khalfA);
        a0.q[1] = *(const uint4*)(srow0 + kb + khalfA + 8);
        a1.q[0] = *(const uint4*)(srow1 + kb + khalfA);
        a1.q[1] = *(const uint4*)(srow1 + kb + khalfA + 8);
#pragma unroll
        for (int jt = 0; jt < 8; ++jt) {
            const int jtG = jh * 8 + jt;
            const _Float16* bbase = Bq + ((size_t)(kt * 16 + jtG) * 32 + lane) * 16;
            Frag16 b;
            b.q[0] = *(const uint4*)(bbase);
            b.q[1] = *(const uint4*)(bbase + 8);
            acc[0][jt] = __builtin_amdgcn_wmma_f32_16x16x32_f16(
                false, a0.h, false, b.h, (short)0, acc[0][jt], false, false);
            acc[1][jt] = __builtin_amdgcn_wmma_f32_16x16x32_f16(
                false, a1.h, false, b.h, (short)0, acc[1][jt], false, false);
        }
    }

    // ---- Epilogue: LDS transpose per 16x16 tile, coalesced I/O, loglik ----
    __shared__ float tiles[8][16][17];    // [wave][t-off][j-off], padded
    __shared__ float red[256];
    float part = 0.f;
    const int mrow = (lane < 16) ? 0 : 8; // D: M = r + mrow

#pragma unroll 1
    for (int h2 = 0; h2 < 2; ++h2) {
        const int tb = tbase0 + h2 * 64;
#pragma unroll 1
        for (int jt = 0; jt < 8; ++jt) {
            const int jbase = (jh * 8 + jt) * 16;
#pragma unroll
            for (int r = 0; r < 8; ++r)
                tiles[wave][r + mrow][jcol] = acc[h2][jt][r];
            __syncthreads();
#pragma unroll
            for (int rr = 0; rr < 8; ++rr) {
                const int jrow = rr * 2 + (lane >> 4);   // 0..15
                const int toff = lane & 15;
                const int tout = tb + 1 + toff;
                const int j    = jbase + jrow;
                const float x  = tiles[wave][toff][jrow];
                const float mu = mu0[j];
                if (tout < Ndim) {
                    const float lam = fmaxf(x, 0.f) + log1pf(expf(-fabsf(x)));
                    const size_t idx = (size_t)j * Ndim + tout;
                    lams1[idx] = lam;                     // 64B-contiguous per j-row
                    lams0[idx] = mu;
                    part += obs[idx] * logf(mu + lam + 1e-5f) - mu - lam;
                }
            }
            __syncthreads();
        }
    }

    red[tid] = part;
    __syncthreads();
    for (int st = 128; st > 0; st >>= 1) {
        if (tid < st) red[tid] += red[tid + st];
        __syncthreads();
    }
    if (tid == 0) partials[blockIdx.x] = red[0];
}

// ---------------------------------------------------------------------------
// Phase 4: t = 0 column (lams1 = 0) + its loglik contribution.
// ---------------------------------------------------------------------------
__global__ void __launch_bounds__(256) hawkes_t0(const float* __restrict__ obs,
                                                 const float* __restrict__ mu0,
                                                 float* __restrict__ lams0,
                                                 float* __restrict__ lams1,
                                                 float* __restrict__ partials) {
    const int k = threadIdx.x;
    const float mu = mu0[k];
    const size_t idx = (size_t)k * Ndim;
    lams0[idx] = mu;
    lams1[idx] = 0.f;
    float part = obs[idx] * logf(mu + 1e-5f) - mu;
    __shared__ float red[256];
    red[k] = part;
    __syncthreads();
    for (int st = 128; st > 0; st >>= 1) {
        if (k < st) red[k] += red[k + st];
        __syncthreads();
    }
    if (k == 0) partials[GEMM_BLOCKS] = red[0];
}

// ---------------------------------------------------------------------------
// Phase 5: deterministic final reduction of partials -> out[0].
// ---------------------------------------------------------------------------
__global__ void __launch_bounds__(256) hawkes_final(const float* __restrict__ partials,
                                                    float* __restrict__ out) {
    float s = 0.f;
    for (int i = threadIdx.x; i < GEMM_BLOCKS + 1; i += 256) s += partials[i];
    __shared__ float red[256];
    red[threadIdx.x] = s;
    __syncthreads();
    for (int st = 128; st > 0; st >>= 1) {
        if (threadIdx.x < st) red[threadIdx.x] += red[threadIdx.x + st];
        __syncthreads();
    }
    if (threadIdx.x == 0) out[0] = red[0];
}

// ---------------------------------------------------------------------------
extern "C" void kernel_launch(void* const* d_in, const int* in_sizes, int n_in,
                              void* d_out, int out_size, void* d_ws, size_t ws_size,
                              hipStream_t stream) {
    const float* obs   = (const float*)d_in[0];   // [256, 65536]
    const float* Beta  = (const float*)d_in[1];   // [256]
    const float* Alpha = (const float*)d_in[2];   // [256, 256]

    float* out   = (float*)d_out;                 // [1 + K*N + K*N]
    float* lams0 = out + 1;
    float* lams1 = out + 1 + (size_t)Kdim * Ndim;

    // Workspace: mu0 (1KB) | Bq f16 (128KB) | partials (~2KB) | S f16 (32MB @256KB)
    char* ws = (char*)d_ws;
    float*    mu0      = (float*)ws;
    _Float16* Bq       = (_Float16*)(ws + 1024);
    float*    partials = (float*)(ws + 1024 + 131072);
    _Float16* S        = (_Float16*)(ws + (1 << 18));

    hawkes_mean      <<<256, 256, 0, stream>>>(obs, mu0);
    hawkes_alpha_pack<<<256, 256, 0, stream>>>(Alpha, Bq);
    hawkes_scan      <<<256, 256, 0, stream>>>(obs, Beta, S);
    hawkes_gemm      <<<GEMM_BLOCKS, 256, 0, stream>>>(S, Bq, mu0, obs,
                                                       lams0, lams1, partials);
    hawkes_t0        <<<1, 256, 0, stream>>>(obs, mu0, lams0, lams1, partials);
    hawkes_final     <<<1, 256, 0, stream>>>(partials, out);
}
